// GINConvLayer_24361054502956
// MI455X (gfx1250) — matrix operand order repaired and checked by reading, not compiled, not run on hardware
//
#include <hip/hip_runtime.h>

#define N_NODES 50000
#define N_EDGES 500000
#define ND 128
#define ED 32
#define EMB 160
#define BN_EPS 1e-5f

#define TROWS 128      // edge rows per block tile (8 waves x 16 rows)
#define KT    5        // K tiles of 32 covering EMB=160
#define JT1   10       // 16-wide col tiles covering EMB=160
#define JT2   8        // 16-wide col tiles covering ND=128
#define WPAD_H 168     // padded LDS row stride in bf16 elems (336 B)
#define WPAD_B 336     // padded LDS row stride in bytes

#define W1L_BYTES (EMB * WPAD_B)          // 53760
#define W2L_BYTES (ND * WPAD_B)           // 43008
#define YT_BYTES  (TROWS * WPAD_B)        // 43008

typedef __attribute__((ext_vector_type(16))) __bf16 v16bf;
typedef __attribute__((ext_vector_type(8)))  float  v8f;
typedef unsigned int u32x4 __attribute__((ext_vector_type(4)));

union ABu { u32x4 u[2]; v16bf v; };

__device__ __forceinline__ v8f wmma_bf16(v16bf a, v16bf b, v8f c) {
  return __builtin_amdgcn_wmma_f32_16x16x32_bf16(
      /*neg_a=*/false, a, /*neg_b=*/false, b,
      /*c_mod=*/(short)0, c, /*reuse_a=*/false, /*reuse_b=*/false);
}

// A fragment from two 8-float global chunks (ISA 16-bit A 16x32 layout:
// lane half 'hi' holds K = hi*8..hi*8+7 in v0-3 and K = 16+hi*8.. in v4-7).
__device__ __forceinline__ v16bf mk_af(const float* c0, const float* c1) {
  float t[16];
  ((float4*)t)[0] = ((const float4*)c0)[0];
  ((float4*)t)[1] = ((const float4*)c0)[1];
  ((float4*)t)[2] = ((const float4*)c1)[0];
  ((float4*)t)[3] = ((const float4*)c1)[1];
  v16bf v;
#pragma unroll
  for (int i = 0; i < 16; i++) v[i] = (__bf16)t[i];
  return v;
}

// Same, but the two 16B chunks come from LDS bf16 (y tile in k3).
__device__ __forceinline__ v16bf mk_af_lds(const __bf16* c0, const __bf16* c1) {
  ABu t;
  t.u[0] = *(const u32x4*)c0;
  t.u[1] = *(const u32x4*)c1;
  return t.v;
}

// Gather this lane's edge row as 5 A fragments straight from global memory.
__device__ __forceinline__ void gather_afrags(v16bf* af, const float* __restrict__ h,
                                              const float* __restrict__ e,
                                              const int* __restrict__ src,
                                              int arow, int hi) {
  bool valid = arow < N_EDGES;
  if (valid) {
    int s = src[arow];
    const float* hrow = h + (size_t)s * ND;
    const float* erow = e + (size_t)arow * ED;
#pragma unroll
    for (int kt = 0; kt < 4; kt++)
      af[kt] = mk_af(hrow + kt * 32 + hi * 8, hrow + kt * 32 + 16 + hi * 8);
    af[4] = mk_af(erow + hi * 8, erow + 16 + hi * 8);
  } else {
    v16bf z;
#pragma unroll
    for (int i = 0; i < 16; i++) z[i] = (__bf16)0.0f;
#pragma unroll
    for (int kt = 0; kt < KT; kt++) af[kt] = z;
  }
}

// Async-copy a row-major bf16 weight matrix (rows x 160) from global into LDS
// with a 336B padded row stride (bank-conflict stagger). 16B chunks, 20/row.
__device__ __forceinline__ void stage_weights_async(const __bf16* __restrict__ Wg,
                                                    unsigned ldsBase, int rows, int tid) {
  int nchunks = rows * 20;
  for (int c = tid; c < nchunks; c += 256) {
    int r = c / 20;
    int q = c - r * 20;
    unsigned loff = ldsBase + (unsigned)(r * WPAD_B + q * 16);
    unsigned long long ga =
        (unsigned long long)(size_t)Wg + (unsigned long long)c * 16ull;
    asm volatile("global_load_async_to_lds_b128 %0, %1, off"
                 :: "v"(loff), "v"(ga) : "memory");
  }
}

__device__ __forceinline__ void wait_async_lds() {
  asm volatile("s_wait_asynccnt 0x0" ::: "memory");
}

// 16x16 tile: C = bias; C += A x B over 5 k-steps, B fragments from padded LDS.
__device__ __forceinline__ v8f tile_gemm_lds(const v16bf* af, const __bf16* Wl,
                                             int jt, int ln, int hi, float bias) {
  v16bf bf[KT];
#pragma unroll
  for (int kt = 0; kt < KT; kt++) {
    ABu t;
    const __bf16* p = &Wl[(jt * 16 + ln) * WPAD_H + kt * 32 + hi * 8];
    t.u[0] = *(const u32x4*)p; t.u[1] = *(const u32x4*)(p + 16);
    bf[kt] = t.v;
  }
  v8f acc;
#pragma unroll
  for (int i = 0; i < 8; i++) acc[i] = bias;
#pragma unroll
  for (int kt = 0; kt < KT; kt++) acc = wmma_bf16(af[kt], bf[kt], acc);
  return acc;
}

// ---------------- kernels ----------------

__global__ __launch_bounds__(256) void k0_convert(const float* __restrict__ W1,
                                                  const float* __restrict__ W2,
                                                  __bf16* __restrict__ W1bf,
                                                  __bf16* __restrict__ W2bf) {
  int i = blockIdx.x * 256 + threadIdx.x;
  if (i < EMB * EMB) W1bf[i] = (__bf16)W1[i];
  if (i < ND * EMB)  W2bf[i] = (__bf16)W2[i];
}

__global__ __launch_bounds__(256) void kz_zero(float* __restrict__ hnew,
                                               float* __restrict__ stats) {
  int i = blockIdx.x * 256 + threadIdx.x;
  if (i < N_NODES * ND) hnew[i] = 0.0f;
  if (i < 2 * EMB) stats[i] = 0.0f;
}

// Phase 1: x = m @ W1^T + b1; per-column sum / sum^2 over all edges.
// W1 served from LDS (async-staged); A gathered straight to registers.
__global__ __launch_bounds__(256) void k1_gemm1_stats(
    const float* __restrict__ h, const float* __restrict__ e,
    const int* __restrict__ src,
    const __bf16* __restrict__ W1bf, const float* __restrict__ b1,
    float* __restrict__ gsum, float* __restrict__ gsq) {
  __shared__ __align__(16) char smem[W1L_BYTES];      // 53760 B
  __bf16* W1l = (__bf16*)smem;
  unsigned ldsBase = (unsigned)(size_t)(void*)smem;

  int tid = threadIdx.x;
  int base = blockIdx.x * TROWS;
  int wave = tid >> 5, lane = tid & 31, ln = lane & 15, hi = lane >> 4;

  stage_weights_async(W1bf, ldsBase, EMB, tid);       // ASYNCcnt traffic

  float b1v[JT1];                                     // per-lane bias, hoisted
#pragma unroll
  for (int jt = 0; jt < JT1; jt++) b1v[jt] = b1[jt * 16 + ln];

  v16bf af[KT];
  gather_afrags(af, h, e, src, base + wave * 16 + ln, hi);  // overlaps async

  wait_async_lds();
  __syncthreads();

  if ((base + TROWS) <= N_EDGES) {                    // fast path: full tile
#pragma unroll
    for (int jt = 0; jt < JT1; jt++) {
      v8f acc = tile_gemm_lds(af, W1l, jt, ln, hi, b1v[jt]);
      float s = 0.0f, s2 = 0.0f;
#pragma unroll
      for (int r2 = 0; r2 < 8; r2++) { float v = acc[r2]; s += v; s2 += v * v; }
      s  += __shfl_xor(s, 16, 32);                    // fold lane halves (same N)
      s2 += __shfl_xor(s2, 16, 32);
      if (hi == 0) {
        atomicAdd(&gsum[jt * 16 + ln], s);
        atomicAdd(&gsq [jt * 16 + ln], s2);
      }
    }
  } else {                                            // last block: masked rows
#pragma unroll
    for (int jt = 0; jt < JT1; jt++) {
      v8f acc = tile_gemm_lds(af, W1l, jt, ln, hi, b1v[jt]);
      float s = 0.0f, s2 = 0.0f;
#pragma unroll
      for (int r2 = 0; r2 < 8; r2++) {
        int grow = base + wave * 16 + hi * 8 + r2;
        float m = (grow < N_EDGES) ? 1.0f : 0.0f;
        float v = acc[r2];
        s += v * m; s2 += v * v * m;
      }
      s  += __shfl_xor(s, 16, 32);
      s2 += __shfl_xor(s2, 16, 32);
      if (hi == 0) {
        atomicAdd(&gsum[jt * 16 + ln], s);
        atomicAdd(&gsq [jt * 16 + ln], s2);
      }
    }
  }
}

// Phase 1.5: fold stats into affine coefs: y = a*x + c.
__global__ __launch_bounds__(192) void k2_stats(const float* __restrict__ gsum,
                                                const float* __restrict__ gsq,
                                                const float* __restrict__ gamma,
                                                const float* __restrict__ beta,
                                                float* __restrict__ cA,
                                                float* __restrict__ cC) {
  int j = threadIdx.x;
  if (j < EMB) {
    float inv = 1.0f / (float)N_EDGES;
    float mu  = gsum[j] * inv;
    float var = gsq[j] * inv - mu * mu;
    float a   = gamma[j] * rsqrtf(var + BN_EPS);
    cA[j] = a;
    cC[j] = beta[j] - mu * a;
  }
}

// Phase 2: recompute GEMM1, BN+ReLU, transpose y through LDS, GEMM2, scatter.
// Both weight matrices async-staged into LDS (139776 B static LDS total).
__global__ __launch_bounds__(256) void k3_gemm2_scatter(
    const float* __restrict__ h, const float* __restrict__ e,
    const int* __restrict__ src, const int* __restrict__ dstIdx,
    const __bf16* __restrict__ W1bf, const float* __restrict__ b1,
    const __bf16* __restrict__ W2bf, const float* __restrict__ b2,
    const float* __restrict__ cA, const float* __restrict__ cC,
    float* __restrict__ hnew) {
  __shared__ __align__(16) char smem[W1L_BYTES + W2L_BYTES + YT_BYTES];
  __bf16* W1l = (__bf16*)smem;
  __bf16* W2l = (__bf16*)(smem + W1L_BYTES);
  __bf16* yt  = (__bf16*)(smem + W1L_BYTES + W2L_BYTES);
  unsigned ldsBase = (unsigned)(size_t)(void*)smem;

  int tid = threadIdx.x;
  int base = blockIdx.x * TROWS;
  int wave = tid >> 5, lane = tid & 31, ln = lane & 15, hi = lane >> 4;

  stage_weights_async(W1bf, ldsBase, EMB, tid);
  stage_weights_async(W2bf, ldsBase + W1L_BYTES, ND, tid);

  float b1v[JT1], cAv[JT1], cCv[JT1], b2v[JT2];       // per-lane, hoisted
#pragma unroll
  for (int jt = 0; jt < JT1; jt++) {
    b1v[jt] = b1[jt * 16 + ln];
    cAv[jt] = cA[jt * 16 + ln];
    cCv[jt] = cC[jt * 16 + ln];
  }
#pragma unroll
  for (int jt = 0; jt < JT2; jt++) b2v[jt] = b2[jt * 16 + ln];

  v16bf af[KT];
  gather_afrags(af, h, e, src, base + wave * 16 + ln, hi);

  int didx[8];                                        // hoist dst gathers
#pragma unroll
  for (int r2 = 0; r2 < 8; r2++) {
    int grow = base + wave * 16 + hi * 8 + r2;
    didx[r2] = (grow < N_EDGES) ? dstIdx[grow] : -1;
  }

  wait_async_lds();
  __syncthreads();

#pragma unroll
  for (int jt = 0; jt < JT1; jt++) {
    v8f acc = tile_gemm_lds(af, W1l, jt, ln, hi, b1v[jt]);
    float a = cAv[jt], c = cCv[jt];
#pragma unroll
    for (int r2 = 0; r2 < 8; r2++) {                  // C layout: M = r2+hi*8, N = ln
      float y = acc[r2] * a + c;
      y = y > 0.0f ? y : 0.0f;
      yt[(wave * 16 + hi * 8 + r2) * WPAD_H + jt * 16 + ln] = (__bf16)y;
    }
  }
  __syncthreads();

  v16bf yf[KT];
#pragma unroll
  for (int kt = 0; kt < KT; kt++) {
    const __bf16* row = &yt[(wave * 16 + ln) * WPAD_H + kt * 32];
    yf[kt] = mk_af_lds(row + hi * 8, row + 16 + hi * 8);
  }

#pragma unroll
  for (int jt = 0; jt < JT2; jt++) {
    v8f acc = tile_gemm_lds(yf, W2l, jt, ln, hi, b2v[jt]);
#pragma unroll
    for (int r2 = 0; r2 < 8; r2++) {
      if (didx[r2] >= 0)
        atomicAdd(&hnew[(size_t)didx[r2] * ND + jt * 16 + ln], acc[r2]);
    }
  }
}

// Epilogue: relu(h_new) in place, relu(e) to second output region.
__global__ __launch_bounds__(256) void k4_epilogue(float* __restrict__ hnew,
                                                   const float* __restrict__ e,
                                                   float* __restrict__ eout) {
  int i = blockIdx.x * 256 + threadIdx.x;
  if (i < N_NODES * ND) { float v = hnew[i]; hnew[i] = v > 0.0f ? v : 0.0f; }
  if (i < N_EDGES * ED) { float v = e[i];    eout[i] = v > 0.0f ? v : 0.0f; }
}

// ---------------- launch ----------------

extern "C" void kernel_launch(void* const* d_in, const int* in_sizes, int n_in,
                              void* d_out, int out_size, void* d_ws, size_t ws_size,
                              hipStream_t stream) {
  (void)in_sizes; (void)n_in; (void)out_size; (void)ws_size;
  const float* h     = (const float*)d_in[0];
  const float* e     = (const float*)d_in[1];
  const int*   src   = (const int*)  d_in[2];
  const int*   dstIx = (const int*)  d_in[3];
  const float* W1    = (const float*)d_in[4];
  const float* b1    = (const float*)d_in[5];
  const float* gamma = (const float*)d_in[6];
  const float* beta  = (const float*)d_in[7];
  const float* W2    = (const float*)d_in[8];
  const float* b2    = (const float*)d_in[9];

  float* hout = (float*)d_out;                       // [N, ND]
  float* eout = hout + (size_t)N_NODES * ND;         // [E, ED]

  char* ws = (char*)d_ws;
  __bf16* W1bf = (__bf16*)(ws);                      // 51200 B
  __bf16* W2bf = (__bf16*)(ws + 51200);              // 40960 B
  float*  gsum = (float*) (ws + 92160);              // 160 f32
  float*  gsq  = (float*) (ws + 92800);              // 160 f32
  float*  cA   = (float*) (ws + 93440);              // 160 f32
  float*  cC   = (float*) (ws + 94080);              // 160 f32

  k0_convert<<<(EMB * EMB + 255) / 256, 256, 0, stream>>>(W1, W2, W1bf, W2bf);
  kz_zero<<<(N_NODES * ND + 255) / 256, 256, 0, stream>>>(hout, gsum);

  int eblocks = (N_EDGES + TROWS - 1) / TROWS;       // 3907
  k1_gemm1_stats<<<eblocks, 256, 0, stream>>>(h, e, src, W1bf, b1, gsum, gsq);
  k2_stats<<<1, 192, 0, stream>>>(gsum, gsq, gamma, beta, cA, cC);
  k3_gemm2_scatter<<<eblocks, 256, 0, stream>>>(h, e, src, dstIx, W1bf, b1,
                                                W2bf, b2, cA, cC, hout);
  k4_epilogue<<<(N_EDGES * ED + 255) / 256, 256, 0, stream>>>(hout, e, eout);
}
